// coAttention_alt_10582799417989
// MI455X (gfx1250) — matrix operand order repaired and checked by reading, not compiled, not run on hardware
//
#include <hip/hip_runtime.h>
#include <hip/hip_bf16.h>
#include <math.h>

// ---- problem constants (match reference) ----
#define BB   256
#define DD   512
#define DKK  100
#define AP   520      // padded LDS row stride in halves (512 + 8) -> avoids bank conflicts
#define WT_ROWS 112   // 7 j-tiles * 16 (rows >= DKK zero-filled)
#define NPMAX 224     // max padded N (13 tiles * 16 = 208, rounded up)

typedef __attribute__((ext_vector_type(16))) _Float16 v16h;
typedef __attribute__((ext_vector_type(8)))  _Float16 v8h;
typedef __attribute__((ext_vector_type(8)))  float    v8f;

union FragU { v16h v; v8h h[2]; };

// One block = one (batch, attend). 8 waves of 32 (wave32).
__global__ __launch_bounds__(256)
void coattend_wmma_kernel(
    const float* __restrict__ ifeat, const float* __restrict__ tfeat,
    const float* __restrict__ gfeat, const float* __restrict__ efeat,
    const float* __restrict__ w0, const float* __restrict__ w1,
    const float* __restrict__ w2, const float* __restrict__ w3,
    const float* __restrict__ wp0, const float* __restrict__ wp1,
    const float* __restrict__ wp2, const float* __restrict__ wp3,
    const float* __restrict__ b0, const float* __restrict__ b1,
    const float* __restrict__ b2, const float* __restrict__ b3,
    float* __restrict__ ws)
{
    const int b    = blockIdx.x;
    const int att  = blockIdx.y;
    const int tid  = threadIdx.x;
    const int lane = tid & 31;
    const int wid  = tid >> 5;

    // select this attend's tensors (softmax is shift-invariant, so the
    // "vec" term of the reference cancels and attends are independent)
    const float* feats; const float* w; const float* wp; const float* bias;
    int N; int wpoff;
    switch (att) {
        case 0:  feats = ifeat; w = w0; wp = wp0; bias = b0; N = 196; wpoff = 0;   break;
        case 1:  feats = tfeat; w = w1; wp = wp1; bias = b1; N = 200; wpoff = DKK; break;
        case 2:  feats = gfeat; w = w2; wp = wp2; bias = b2; N = 200; wpoff = DKK; break;
        default: feats = efeat; w = w3; wp = wp3; bias = b3; N = 50;  wpoff = DKK; break;
    }
    feats += (size_t)b * (size_t)N * DD;
    const int numTiles = (N + 15) >> 4;

    extern __shared__ unsigned char smem_raw[];
    _Float16* wt    = (_Float16*)smem_raw;          // [WT_ROWS][AP] f16, w transposed (j-major)
    _Float16* atile = wt + WT_ROWS * AP;            // [16][AP] f16, current 16-row feats tile
    float*    score = (float*)(atile + 16 * AP);    // [NPMAX]
    float*    pbuf  = score + NPMAX;                // [NPMAX]
    float*    red   = pbuf + NPMAX;                 // [8]

    // ---- stage W transposed to f16 LDS: wt[j*AP + k] = w[k*DKK + j] ----
    for (int idx = tid; idx < DD * DKK; idx += 256) {
        int k = idx / DKK, j = idx - k * DKK;       // coalesced global read
        wt[j * AP + k] = (_Float16)w[idx];
    }
    for (int idx = tid; idx < (WT_ROWS - DKK) * AP; idx += 256)
        wt[DKK * AP + idx] = (_Float16)0.f;         // zero pad j = 100..111

    // ---- score init with bias ----
    for (int i = tid; i < NPMAX; i += 256)
        score[i] = (i < N) ? bias[i] : 0.f;

    // per-lane projection weight wf[j] (constant across row tiles)
    const int jt   = wid;                           // waves 0..6 own one 16-wide DK tile
    const int jcol = jt * 16 + (lane & 15);
    float wfv = 0.f;
    if (jt < 7 && jcol < DKK) wfv = wp[wpoff + jcol];

    __syncthreads();

    const int nlocal_base = (lane >> 4) << 3;       // C layout: lanes 16..31 hold rows +8
    const float4* f4 = (const float4*)feats;

    for (int mt = 0; mt < numTiles; ++mt) {
        // ---- stage A tile (16 rows x 512 k) fp32 -> f16, coalesced ----
        for (int idx = tid; idx < 16 * (DD / 4); idx += 256) {
            int r = idx >> 7, k4 = idx & 127;
            int row = mt * 16 + r;
            float4 v = make_float4(0.f, 0.f, 0.f, 0.f);
            if (row < N) v = f4[row * (DD / 4) + k4];
            int prow = row + 16;
            if (prow < N) __builtin_prefetch(&f4[prow * (DD / 4) + k4], 0, 1);
            _Float16* dst = atile + r * AP + k4 * 4;
            dst[0] = (_Float16)v.x; dst[1] = (_Float16)v.y;
            dst[2] = (_Float16)v.z; dst[3] = (_Float16)v.w;
        }
        __syncthreads();

        if (wid < 7) {
            // C(mt, jt) = A(16x512) x W(512x16), f16 in / f32 acc
            v8f acc = {0.f, 0.f, 0.f, 0.f, 0.f, 0.f, 0.f, 0.f};
            const int arow = lane & 15;
            const int aoff = (lane < 16) ? 0 : 8;   // A: lane-half K split 0..7 / 8..15 (+16..)
            const int boff = (lane < 16) ? 0 : 8;   // B: lane-half K split 0..7 / 8..15 (+16..)
            const _Float16* abase = atile + arow * AP;
            const _Float16* bbase = wt + jcol * AP;
            #pragma unroll
            for (int ks = 0; ks < 16; ++ks) {
                const int k0 = ks * 32;
                FragU fa, fb;
                fa.h[0] = *(const v8h*)(abase + k0 + aoff);
                fa.h[1] = *(const v8h*)(abase + k0 + aoff + 16);
                fb.h[0] = *(const v8h*)(bbase + k0 + boff);
                fb.h[1] = *(const v8h*)(bbase + k0 + boff + 16);
                acc = __builtin_amdgcn_wmma_f32_16x16x32_f16(
                        false, fa.v, false, fb.v, (short)0, acc, false, false);
            }
            // score[n] += sum_j tanh(sf[n,j]) * wf[j]
            #pragma unroll
            for (int r = 0; r < 8; ++r) {
                float v = tanhf(acc[r]) * wfv;
                v += __shfl_xor(v, 1, 32);
                v += __shfl_xor(v, 2, 32);
                v += __shfl_xor(v, 4, 32);
                v += __shfl_xor(v, 8, 32);          // reduce over 16-lane half (one j-tile)
                if ((lane & 15) == 0)
                    atomicAdd(&score[mt * 16 + nlocal_base + r], v);
            }
        }
        __syncthreads();
    }

    // ---- block softmax over score[0..N) ----
    float sv = (tid < N) ? score[tid] : -3.402823466e38f;
    float m = sv;
    m = fmaxf(m, __shfl_xor(m, 1, 32));
    m = fmaxf(m, __shfl_xor(m, 2, 32));
    m = fmaxf(m, __shfl_xor(m, 4, 32));
    m = fmaxf(m, __shfl_xor(m, 8, 32));
    m = fmaxf(m, __shfl_xor(m, 16, 32));
    if (lane == 0) red[wid] = m;
    __syncthreads();
    float mx = red[0];
    #pragma unroll
    for (int i = 1; i < 8; ++i) mx = fmaxf(mx, red[i]);
    __syncthreads();                                 // red reused below
    float e = (tid < N) ? expf(sv - mx) : 0.f;
    float s = e;
    s += __shfl_xor(s, 1, 32);
    s += __shfl_xor(s, 2, 32);
    s += __shfl_xor(s, 4, 32);
    s += __shfl_xor(s, 8, 32);
    s += __shfl_xor(s, 16, 32);
    if (lane == 0) red[wid] = s;
    __syncthreads();
    float tot = 0.f;
    #pragma unroll
    for (int i = 0; i < 8; ++i) tot += red[i];
    if (tid < N) pbuf[tid] = e * (1.f / tot);
    __syncthreads();

    // ---- V[d] = sum_n p[n] * feats[n,d] : fully coalesced streaming pass ----
    float a0 = 0.f, a1 = 0.f;
    const int d0 = tid, d1 = tid + 256;
    #pragma unroll 4
    for (int n = 0; n < N; ++n) {
        float pn = pbuf[n];
        const float* row = feats + (size_t)n * DD;
        a0 = fmaf(pn, row[d0], a0);
        a1 = fmaf(pn, row[d1], a1);
    }
    float* wsv = ws + ((size_t)att * BB + b) * DD;
    wsv[d0] = a0;
    wsv[d1] = a1;
}

// out = V0 + V1 + 0.5*(V2 + V3)
__global__ __launch_bounds__(256)
void coattend_combine_kernel(const float* __restrict__ ws, float* __restrict__ out)
{
    const int tot = BB * DD;
    int i = blockIdx.x * 256 + threadIdx.x;
    if (i < tot)
        out[i] = ws[i] + ws[tot + i] + 0.5f * (ws[2 * tot + i] + ws[3 * tot + i]);
}

extern "C" void kernel_launch(void* const* d_in, const int* in_sizes, int n_in,
                              void* d_out, int out_size, void* d_ws, size_t ws_size,
                              hipStream_t stream) {
    (void)in_sizes; (void)n_in; (void)out_size; (void)ws_size;
    const float* ifeat = (const float*)d_in[0];
    const float* tfeat = (const float*)d_in[1];
    const float* gfeat = (const float*)d_in[2];
    const float* efeat = (const float*)d_in[3];
    // per-attend: w_feat is w_vi0 for attend0, w_vt{i} for attends 1..3
    const float* w0  = (const float*)d_in[4];   // w_vi0
    const float* wp0 = (const float*)d_in[6];
    const float* b0  = (const float*)d_in[7];
    const float* w1  = (const float*)d_in[9];   // w_vt1
    const float* wp1 = (const float*)d_in[10];
    const float* b1  = (const float*)d_in[11];
    const float* w2  = (const float*)d_in[13];  // w_vt2
    const float* wp2 = (const float*)d_in[14];
    const float* b2  = (const float*)d_in[15];
    const float* w3  = (const float*)d_in[17];  // w_vt3
    const float* wp3 = (const float*)d_in[18];
    const float* b3  = (const float*)d_in[19];

    float* ws  = (float*)d_ws;   // 4 * 256 * 512 floats = 2 MB
    float* out = (float*)d_out;

    const size_t smem = (size_t)(WT_ROWS * AP + 16 * AP) * sizeof(_Float16)
                      + (size_t)(2 * NPMAX + 8) * sizeof(float);

    coattend_wmma_kernel<<<dim3(BB, 4), 256, smem, stream>>>(
        ifeat, tfeat, gfeat, efeat,
        w0, w1, w2, w3, wp0, wp1, wp2, wp3, b0, b1, b2, b3, ws);

    coattend_combine_kernel<<<(BB * DD + 255) / 256, 256, 0, stream>>>(ws, out);
}
